// CrossCovarianceAttn_17566416241047
// MI455X (gfx1250) — matrix operand order repaired and checked by reading, not compiled, not run on hardware
//
#include <hip/hip_runtime.h>

// ---------------------------------------------------------------------------
// CrossCovarianceAttn for MI455X (gfx1250, wave32, WMMA bf16 w/ f32 accum)
// B=8, N=1024, C=768, H=8, hd=96
// Pipeline:
//   0a) x fp32 -> bf16                  (xb  [8192][768])
//   0b) w_qkv^T  fp32 -> bf16           (wqt [2304][768])
//   0c) w_proj^T fp32 -> bf16           (wpt [768][768])
//   1)  qkv GEMM + fused L2-norm(+temp) -> Qn,Kn [bh][tok][96], Vt [bh][96][tok]
//   2)  flash attention per (bh, 64-row block), K/V staged via
//       GLOBAL_LOAD_ASYNC_TO_LDS_B128 double buffering -> Oa [8192][768] bf16
//   3)  proj GEMM + bias -> d_out fp32
// ---------------------------------------------------------------------------

typedef __attribute__((ext_vector_type(16))) __bf16 v16bf;
typedef __attribute__((ext_vector_type(8)))  __bf16 v8bf;
typedef __attribute__((ext_vector_type(8)))  float  v8f;

#define B_   8
#define N_   1024
#define C_   768
#define H_   8
#define HD_  96

// ---- workspace layout (bytes) ----
static constexpr size_t OFF_XB  = 0;                               // [8192][768] bf16
static constexpr size_t OFF_WQT = OFF_XB  + (size_t)8192*768*2;    // [2304][768] bf16
static constexpr size_t OFF_WPT = OFF_WQT + (size_t)2304*768*2;    // [768][768]  bf16
static constexpr size_t OFF_QN  = OFF_WPT + (size_t)768*768*2;     // [64][1024][96] bf16
static constexpr size_t OFF_KN  = OFF_QN  + (size_t)64*1024*96*2;
static constexpr size_t OFF_VT  = OFF_KN  + (size_t)64*1024*96*2;  // [64][96][1024] bf16
static constexpr size_t OFF_OA  = OFF_VT  + (size_t)64*96*1024*2;  // [8192][768] bf16

__device__ __forceinline__ v8f wmma_bf16(v16bf a, v16bf b, v8f c) {
  // v_wmma_f32_16x16x32_bf16: D = A(16x32) * B(32x16) + C(16x16 f32)
  return __builtin_amdgcn_wmma_f32_16x16x32_bf16(
      /*neg_a=*/false, a, /*neg_b=*/false, b,
      /*c_mod=*/(short)0, c, /*reuse_a=*/false, /*reuse_b=*/false);
}

// A fragment (16x32 bf16) from a row-major operand: lane L holds row (L&15),
// K = 8*(L>>4)+[0..7] and +16..+23 (two contiguous 16B chunks).
__device__ __forceinline__ v16bf load_a_frag(const __bf16* base, int ld, int lane) {
  int row = lane & 15;
  int kb  = (lane >> 4) * 8;
  const __bf16* p = base + (size_t)row * ld + kb;
  v8bf lo = *(const v8bf*)(p);
  v8bf hi = *(const v8bf*)(p + 16);
  return __builtin_shufflevector(lo, hi, 0,1,2,3,4,5,6,7,8,9,10,11,12,13,14,15);
}

// B fragment (32x16 bf16) from a TRANSPOSED (i.e. [n][k] row-major) operand:
// lane L holds column n=(L&15): k = kb0+8*(L>>4)+[0..7] and +16..+23 (two 16B chunks).
__device__ __forceinline__ v16bf load_b_frag(const __bf16* baseT, int ld,
                                             int col0, int kb0, int lane) {
  const __bf16* p = baseT + (size_t)(col0 + (lane & 15)) * ld
                          + kb0 + (lane >> 4) * 8;
  v8bf lo = *(const v8bf*)(p);
  v8bf hi = *(const v8bf*)(p + 16);
  return __builtin_shufflevector(lo, hi, 0,1,2,3,4,5,6,7,8,9,10,11,12,13,14,15);
}

// Async copy of one 16B chunk per lane: global -> LDS, tracked by ASYNCcnt.
// The LDS destination VGPR carries the low 32 bits of the flat shared address,
// which per ISA 10.2 is the LDS byte
// offset.
__device__ __forceinline__ void async_load_b128(const void* gptr, void* lptr) {
  unsigned lds_off = (unsigned)(uintptr_t)lptr;
  asm volatile("global_load_async_to_lds_b128 %0, %1, off"
               :: "v"(lds_off), "v"(gptr)
               : "memory");
}

__device__ __forceinline__ void wait_asynccnt0() {
  asm volatile("s_wait_asynccnt 0x0" ::: "memory");
}

// ---------------------------------------------------------------------------
// prep kernels
// ---------------------------------------------------------------------------
__global__ void cvt_f32_bf16_kernel(const float* __restrict__ src,
                                    __bf16* __restrict__ dst, int n) {
  int i = blockIdx.x * blockDim.x + threadIdx.x;
  if (i < n) dst[i] = (__bf16)src[i];
}

// w [rows][cols] fp32 -> wt [cols][rows] bf16
__global__ void transpose_cvt_kernel(const float* __restrict__ w,
                                     __bf16* __restrict__ wt,
                                     int rows, int cols) {
  int i = blockIdx.x * blockDim.x + threadIdx.x;
  if (i < rows * cols) {
    int r = i / cols, c = i - r * cols;
    wt[(size_t)c * rows + r] = (__bf16)w[i];
  }
}

// ---------------------------------------------------------------------------
// 1) qkv = x @ w_qkv, fused L2-normalize (q,k) + temperature fold into q.
//    grid = (128 rowblocks of 64 tokens, 24 = three*8+h), block = 128 (4 waves)
// ---------------------------------------------------------------------------
__global__ __launch_bounds__(128) void qkv_gemm_norm_kernel(
    const __bf16* __restrict__ xb,    // [8192][768]
    const __bf16* __restrict__ wqt,   // [2304][768]  (= w_qkv^T)
    const float*  __restrict__ temp,  // [8]
    __bf16* __restrict__ Qn,          // [64][1024][96]
    __bf16* __restrict__ Kn,          // [64][1024][96]
    __bf16* __restrict__ Vt)          // [64][96][1024]
{
  const int lane = threadIdx.x & 31;
  const int wave = threadIdx.x >> 5;
  const int rb   = blockIdx.x;            // 0..127 over B*N/64
  const int cg   = blockIdx.y;            // 0..23
  const int three = cg >> 3, h = cg & 7;
  const int row0 = rb * 64 + wave * 16;   // token-row in [0, 8192)
  const int col0 = three * C_ + h * HD_;  // column in [0, 2304)

  v8f acc[6];
  #pragma unroll
  for (int j = 0; j < 6; ++j)
    #pragma unroll
    for (int r = 0; r < 8; ++r) acc[j][r] = 0.0f;

  for (int k0 = 0; k0 < C_; k0 += 32) {
    v16bf a = load_a_frag(xb + (size_t)row0 * C_ + k0, C_, lane);
    #pragma unroll
    for (int j = 0; j < 6; ++j) {
      v16bf b = load_b_frag(wqt, C_, col0 + 16 * j, k0, lane);
      acc[j] = wmma_bf16(a, b, acc[j]);
    }
  }

  const int hi = lane >> 4, c = lane & 15;
  const int b_ = row0 >> 10;          // batch
  const int n0 = row0 & (N_ - 1);     // token within sequence
  const int bh = b_ * H_ + h;

  if (three == 2) {
    // V stored transposed [bh][hd][token]; accumulator r runs over 8
    // consecutive tokens -> pack one 16B store per (lane, j).
    #pragma unroll
    for (int j = 0; j < 6; ++j) {
      v8bf pk;
      #pragma unroll
      for (int r = 0; r < 8; ++r) pk[r] = (__bf16)acc[j][r];
      __bf16* dst = Vt + ((size_t)bh * HD_ + (c + 16 * j)) * N_ + n0 + 8 * hi;
      *(v8bf*)dst = pk;
    }
  } else {
    // L2 norm over the 96 head-dim values of each row (spread across the
    // 16 lanes of this lane's half-wave + 6 accumulators).
    const float ts = (three == 0) ? temp[h] : 1.0f;
    __bf16* dst0 = ((three == 0) ? Qn : Kn) + ((size_t)bh * N_ + n0) * HD_;
    #pragma unroll
    for (int r = 0; r < 8; ++r) {
      float ssq = 0.0f;
      #pragma unroll
      for (int j = 0; j < 6; ++j) ssq += acc[j][r] * acc[j][r];
      #pragma unroll
      for (int m = 1; m < 16; m <<= 1) ssq += __shfl_xor(ssq, m, 32);
      float scale = ts / fmaxf(sqrtf(ssq), 1e-12f);
      #pragma unroll
      for (int j = 0; j < 6; ++j)
        dst0[(size_t)(r + 8 * hi) * HD_ + c + 16 * j] = (__bf16)(acc[j][r] * scale);
    }
  }
}

// ---------------------------------------------------------------------------
// 2) flash attention: per (bh, 64-row block). Wave owns 16 query rows.
//    K tile [32 tok][96] and V tile [96][32 tok] staged into LDS once per
//    workgroup via async-to-LDS loads, double buffered against compute.
//    S = Q@K^T (6 wmma / 32 cols), online softmax, O += P@V (6 wmma).
// ---------------------------------------------------------------------------
#define KS_ELEMS (32 * HD_)          // 3072 bf16 = 6 KB
#define VS_ELEMS (HD_ * 32)          // 3072 bf16 = 6 KB
#define BUF_ELEMS (KS_ELEMS + VS_ELEMS)

__device__ __forceinline__ void issue_kv_loads(
    const __bf16* __restrict__ kb, const __bf16* __restrict__ vtb,
    int t0, __bf16* buf, int tid)
{
  // K tile: 32 rows x 192B = 384 x 16B chunks; 128 threads x 3
  #pragma unroll
  for (int i = 0; i < 3; ++i) {
    int q = i * 128 + tid;
    int row = q / 12, cj = q - row * 12;
    async_load_b128(kb + (size_t)(t0 + row) * HD_ + cj * 8,
                    buf + row * HD_ + cj * 8);
  }
  // V tile (transposed src): 96 rows x 64B = 384 x 16B chunks; 128 threads x 3
  #pragma unroll
  for (int i = 0; i < 3; ++i) {
    int q = i * 128 + tid;
    int row = q >> 2, cj = q & 3;
    async_load_b128(vtb + (size_t)row * N_ + t0 + cj * 8,
                    buf + KS_ELEMS + row * 32 + cj * 8);
  }
}

__global__ __launch_bounds__(128) void attention_kernel(
    const __bf16* __restrict__ Qn,   // [64][1024][96] (temp folded in)
    const __bf16* __restrict__ Kn,   // [64][1024][96]
    const __bf16* __restrict__ Vt,   // [64][96][1024]
    __bf16* __restrict__ Oa)         // [8192][768]
{
  __shared__ __align__(16) __bf16 kv[2][BUF_ELEMS];     // 24 KB double buffer
  __shared__ __align__(16) __bf16 plds[4][16][32];      // per-wave P bounce

  const int tid  = threadIdx.x;
  const int lane = tid & 31;
  const int wave = tid >> 5;
  const int bh   = blockIdx.y;                 // 0..63
  const int rb   = blockIdx.x;                 // 0..15
  const int row0 = rb * 64 + wave * 16;        // query row in [0,1024)
  const int hi = lane >> 4, c = lane & 15;

  const __bf16* qb  = Qn + (size_t)bh * N_ * HD_;
  const __bf16* kb  = Kn + (size_t)bh * N_ * HD_;
  const __bf16* vtb = Vt + (size_t)bh * HD_ * N_;

  // Q fragments for this wave's 16 rows (hd = 96 -> 3 K-chunks of 32)
  v16bf qa[3];
  #pragma unroll
  for (int ck = 0; ck < 3; ++ck)
    qa[ck] = load_a_frag(qb + (size_t)row0 * HD_ + 32 * ck, HD_, lane);

  v8f o[6];
  float mrow[8], lrow[8];
  #pragma unroll
  for (int j = 0; j < 6; ++j)
    #pragma unroll
    for (int r = 0; r < 8; ++r) o[j][r] = 0.0f;
  #pragma unroll
  for (int r = 0; r < 8; ++r) { mrow[r] = -3.0e38f; lrow[r] = 0.0f; }

  // prologue: stage first K/V block
  issue_kv_loads(kb, vtb, 0, kv[0], tid);
  wait_asynccnt0();
  __syncthreads();

  int cur = 0;
  for (int t0 = 0; t0 < N_; t0 += 32) {
    // prefetch next block into the other buffer (overlaps with compute)
    if (t0 + 32 < N_)
      issue_kv_loads(kb, vtb, t0 + 32, kv[cur ^ 1], tid);

    const __bf16* Ks = kv[cur];              // [32 tok][96]
    const __bf16* Vs = kv[cur] + KS_ELEMS;   // [96][32 tok]

    // --- scores for 32 key columns: two 16x16 tiles ---
    v8f s0, s1;
    #pragma unroll
    for (int r = 0; r < 8; ++r) { s0[r] = 0.0f; s1[r] = 0.0f; }
    #pragma unroll
    for (int ck = 0; ck < 3; ++ck) {
      v16bf b0 = load_b_frag(Ks, HD_, 0,  32 * ck, lane);
      v16bf b1 = load_b_frag(Ks, HD_, 16, 32 * ck, lane);
      s0 = wmma_bf16(qa[ck], b0, s0);
      s1 = wmma_bf16(qa[ck], b1, s1);
    }

    // --- online softmax update (row stats across 16-lane halves) ---
    float alpha[8];
    #pragma unroll
    for (int r = 0; r < 8; ++r) {
      float mx = fmaxf(s0[r], s1[r]);
      #pragma unroll
      for (int m = 1; m < 16; m <<= 1) mx = fmaxf(mx, __shfl_xor(mx, m, 32));
      float mnew = fmaxf(mrow[r], mx);
      float p0 = __expf(s0[r] - mnew);
      float p1 = __expf(s1[r] - mnew);
      s0[r] = p0; s1[r] = p1;
      float ps = p0 + p1;
      #pragma unroll
      for (int m = 1; m < 16; m <<= 1) ps += __shfl_xor(ps, m, 32);
      alpha[r] = __expf(mrow[r] - mnew);
      lrow[r]  = lrow[r] * alpha[r] + ps;
      mrow[r]  = mnew;
    }

    // --- P: accumulator layout -> A layout via per-wave LDS tile ---
    #pragma unroll
    for (int r = 0; r < 8; ++r) {
      plds[wave][r + 8 * hi][c]      = (__bf16)s0[r];
      plds[wave][r + 8 * hi][c + 16] = (__bf16)s1[r];
    }
    asm volatile("s_wait_dscnt 0x0" ::: "memory");
    v16bf pa = load_a_frag(&plds[wave][0][0], 32, lane);

    // --- rescale O, accumulate P @ V ---
    #pragma unroll
    for (int j = 0; j < 6; ++j) {
      #pragma unroll
      for (int r = 0; r < 8; ++r) o[j][r] *= alpha[r];
      v16bf vb = load_b_frag(Vs, 32, 16 * j, 0, lane);
      o[j] = wmma_bf16(pa, vb, o[j]);
    }

    // next buffer landed; everyone done reading current buffer
    wait_asynccnt0();
    __syncthreads();
    cur ^= 1;
  }

  // --- finalize and scatter to [B][N][C] bf16 ---
  const int b_ = bh >> 3, h_ = bh & 7;
  __bf16* dst = Oa + ((size_t)b_ * N_ + row0) * C_ + h_ * HD_;
  #pragma unroll
  for (int r = 0; r < 8; ++r) {
    float inv = 1.0f / lrow[r];
    #pragma unroll
    for (int j = 0; j < 6; ++j)
      dst[(size_t)(r + 8 * hi) * C_ + c + 16 * j] = (__bf16)(o[j][r] * inv);
  }
}

// ---------------------------------------------------------------------------
// 3) out = Oa @ w_proj + b_proj, fp32 output.
//    grid = (128 rowblocks, 8 colgroups of 96), block = 128 (4 waves)
// ---------------------------------------------------------------------------
__global__ __launch_bounds__(128) void proj_gemm_kernel(
    const __bf16* __restrict__ Oa,   // [8192][768]
    const __bf16* __restrict__ wpt,  // [768][768] (= w_proj^T)
    const float*  __restrict__ bias, // [768]
    float* __restrict__ out)         // [8192][768]
{
  const int lane = threadIdx.x & 31;
  const int wave = threadIdx.x >> 5;
  const int row0 = blockIdx.x * 64 + wave * 16;
  const int col0 = blockIdx.y * 96;

  v8f acc[6];
  #pragma unroll
  for (int j = 0; j < 6; ++j)
    #pragma unroll
    for (int r = 0; r < 8; ++r) acc[j][r] = 0.0f;

  for (int k0 = 0; k0 < C_; k0 += 32) {
    v16bf a = load_a_frag(Oa + (size_t)row0 * C_ + k0, C_, lane);
    #pragma unroll
    for (int j = 0; j < 6; ++j) {
      v16bf b = load_b_frag(wpt, C_, col0 + 16 * j, k0, lane);
      acc[j] = wmma_bf16(a, b, acc[j]);
    }
  }

  const int hi = lane >> 4, c = lane & 15;
  #pragma unroll
  for (int r = 0; r < 8; ++r) {
    int tok = row0 + r + 8 * hi;
    #pragma unroll
    for (int j = 0; j < 6; ++j) {
      int col = col0 + c + 16 * j;
      out[(size_t)tok * C_ + col] = acc[j][r] + bias[col];
    }
  }
}

// ---------------------------------------------------------------------------
extern "C" void kernel_launch(void* const* d_in, const int* in_sizes, int n_in,
                              void* d_out, int out_size, void* d_ws, size_t ws_size,
                              hipStream_t stream) {
  const float* x      = (const float*)d_in[0];   // [8,1024,768]
  const float* w_qkv  = (const float*)d_in[1];   // [768,2304]
  const float* temp   = (const float*)d_in[2];   // [8,1,1]
  const float* w_proj = (const float*)d_in[3];   // [768,768]
  const float* b_proj = (const float*)d_in[4];   // [768]

  char* ws = (char*)d_ws;
  __bf16* xb  = (__bf16*)(ws + OFF_XB);
  __bf16* wqt = (__bf16*)(ws + OFF_WQT);
  __bf16* wpt = (__bf16*)(ws + OFF_WPT);
  __bf16* Qn  = (__bf16*)(ws + OFF_QN);
  __bf16* Kn  = (__bf16*)(ws + OFF_KN);
  __bf16* Vt  = (__bf16*)(ws + OFF_VT);
  __bf16* Oa  = (__bf16*)(ws + OFF_OA);

  const int nx = B_ * N_ * C_;  // 6,291,456
  cvt_f32_bf16_kernel<<<(nx + 255) / 256, 256, 0, stream>>>(x, xb, nx);

  const int nwq = C_ * 3 * C_;  // 1,769,472
  transpose_cvt_kernel<<<(nwq + 255) / 256, 256, 0, stream>>>(w_qkv, wqt, C_, 3 * C_);

  const int nwp = C_ * C_;      // 589,824
  transpose_cvt_kernel<<<(nwp + 255) / 256, 256, 0, stream>>>(w_proj, wpt, C_, C_);

  qkv_gemm_norm_kernel<<<dim3(128, 24), 128, 0, stream>>>(xb, wqt, temp, Qn, Kn, Vt);

  attention_kernel<<<dim3(16, 64), 128, 0, stream>>>(Qn, Kn, Vt, Oa);

  proj_gemm_kernel<<<dim3(128, 8), 128, 0, stream>>>(Oa, wpt, b_proj, (float*)d_out);
}
